// GPT_3959959847300
// MI455X (gfx1250) — matrix operand order repaired and checked by reading, not compiled, not run on hardware
//
#include <hip/hip_runtime.h>
#include <hip/hip_bf16.h>

// ---------------- problem constants (from reference) ----------------
#define Lc 4
#define Hc 8
#define Cc 512
#define Vc 50257
#define Vpad 50304     /* Vc rounded up to multiple of 64 for unguarded loads */
#define Tc 1024
#define Bc 2
#define Rc 8
#define Mc (Bc * Tc)   /* 2048 rows of activations */
#define Dh 64          /* head dim */
#define LORA_SCALE 4.0f
#define QSCALE 0.125f  /* 1/sqrt(64) folded into q */

typedef __attribute__((ext_vector_type(16))) __bf16 v16bf;
typedef __attribute__((ext_vector_type(8)))  float  v8f;

union BFrag { v16bf v; uint4 q[2]; };

static __device__ __forceinline__ v8f wmma_bf16(BFrag a, BFrag b, v8f c) {
  // D = A(16x32 bf16) * B(32x16 bf16) + C(16x16 f32)
  return __builtin_amdgcn_wmma_f32_16x16x32_bf16(
      false, a.v, false, b.v, (short)0, c, false, false);
}

static __device__ __forceinline__ v8f zero8() {
  v8f z = {0.f, 0.f, 0.f, 0.f, 0.f, 0.f, 0.f, 0.f};
  return z;
}

// ---------------- f32 -> bf16 conversion with zero padding ----------------
__global__ void cvt_bf16_kernel(__bf16* __restrict__ dst,
                                const float* __restrict__ src, int n, int npad) {
  int i = blockIdx.x * blockDim.x + threadIdx.x;
  if (i >= npad) return;
  dst[i] = (i < n) ? (__bf16)src[i] : (__bf16)0.0f;
}

// ---------------- embedding: x = wte[idx] + wpe[t] ----------------
__global__ void embed_kernel(float* __restrict__ x, const int* __restrict__ idx,
                             const float* __restrict__ wte,
                             const float* __restrict__ wpe) {
  int i = blockIdx.x * blockDim.x + threadIdx.x;
  if (i >= Mc * Cc) return;
  int m = i / Cc, c = i % Cc;
  int tok = idx[m];
  x[i] = wte[(size_t)tok * Cc + c] + wpe[(size_t)(m % Tc) * Cc + c];
}

// ---------------- LayerNorm (f32 in, bf16 out), one block per row ----------------
__global__ __launch_bounds__(256) void ln_kernel(__bf16* __restrict__ out,
                                                 const float* __restrict__ x,
                                                 const float* __restrict__ g,
                                                 const float* __restrict__ b) {
  __shared__ float ssum[256], ssq[256];
  int row = blockIdx.x, tid = threadIdx.x;
  const float* xr = x + (size_t)row * Cc;
  float a0 = xr[tid], a1 = xr[tid + 256];
  ssum[tid] = a0 + a1;
  ssq[tid] = a0 * a0 + a1 * a1;
  __syncthreads();
  for (int s = 128; s > 0; s >>= 1) {
    if (tid < s) { ssum[tid] += ssum[tid + s]; ssq[tid] += ssq[tid + s]; }
    __syncthreads();
  }
  float mean = ssum[0] * (1.0f / Cc);
  float var = ssq[0] * (1.0f / Cc) - mean * mean;
  float rstd = rsqrtf(var + 1e-5f);
  __bf16* orow = out + (size_t)row * Cc;
  orow[tid]       = (__bf16)((a0 - mean) * rstd * g[tid] + b[tid]);
  orow[tid + 256] = (__bf16)((a1 - mean) * rstd * g[tid + 256] + b[tid + 256]);
}

// ---------------- WMMA GEMM: out[M,N] f32 = A[M,K]bf16 . W[Npad,K]bf16^T (+bias)
// block = 128 threads (4 waves); each wave computes a 32(M) x 64(N) slab.
// W must have ceil(N/64)*64 rows (pad rows zero) -> no load guards in the hot loop.
__global__ __launch_bounds__(128) void wmma_gemm_kernel(
    float* __restrict__ out, const __bf16* __restrict__ A,
    const __bf16* __restrict__ W, const float* __restrict__ bias,
    int M, int N, int K, int nt_store) {
  const int lane = threadIdx.x & 31;
  const int wave = threadIdx.x >> 5;
  const int m0 = blockIdx.y * 128 + wave * 32;
  const int n0 = blockIdx.x * 64;
  const int r = lane & 15;   // A: row / B,C: column within 16-tile
  const int sel = lane >> 4; // K half-select (A,B) / M half-select (C)

  const __bf16* arow0 = A + (size_t)(m0 + r) * K;
  const __bf16* arow1 = arow0 + (size_t)16 * K;
  v8f acc[2][4] = {{zero8(), zero8(), zero8(), zero8()},
                   {zero8(), zero8(), zero8(), zero8()}};

  for (int k0 = 0; k0 < K; k0 += 32) {
    BFrag a0, a1;
    a0.q[0] = *(const uint4*)(arow0 + k0 + sel * 8);
    a0.q[1] = *(const uint4*)(arow0 + k0 + 16 + sel * 8);
    a1.q[0] = *(const uint4*)(arow1 + k0 + sel * 8);
    a1.q[1] = *(const uint4*)(arow1 + k0 + 16 + sel * 8);
    const __bf16* wp = W + (size_t)(n0 + r) * K + k0;
#pragma unroll
    for (int nt = 0; nt < 4; ++nt) {
      BFrag bw;
      const __bf16* wq = wp + (size_t)nt * 16 * K;
      bw.q[0] = *(const uint4*)(wq + sel * 8);
      bw.q[1] = *(const uint4*)(wq + 16 + sel * 8);
      acc[0][nt] = wmma_bf16(a0, bw, acc[0][nt]);
      acc[1][nt] = wmma_bf16(a1, bw, acc[1][nt]);
    }
  }
  // C layout: lane<16 -> col=lane, rows v=0..7; lane>=16 -> col=lane-16, rows 8+v
#pragma unroll
  for (int mt = 0; mt < 2; ++mt) {
    int mrow = m0 + mt * 16 + sel * 8;
#pragma unroll
    for (int nt = 0; nt < 4; ++nt) {
      int col = n0 + nt * 16 + r;
      if (col >= N) continue;
      float bv = bias ? bias[col] : 0.0f;
      float* op = out + (size_t)mrow * N + col;
      if (nt_store) {
#pragma unroll
        for (int v = 0; v < 8; ++v)
          __builtin_nontemporal_store(acc[mt][nt][v] + bv, op + (size_t)v * N);
      } else {
#pragma unroll
        for (int v = 0; v < 8; ++v) op[(size_t)v * N] = acc[mt][nt][v] + bv;
      }
    }
  }
}

// ---------------- LoRA rank-8 projection: t[M,R] = h_bf16 . lA[R,K]^T ----------
__global__ void lora_t_kernel(float* __restrict__ t, const __bf16* __restrict__ h,
                              const float* __restrict__ lA) {
  int i = blockIdx.x * blockDim.x + threadIdx.x;
  if (i >= Mc * Rc) return;
  int m = i >> 3, rr = i & 7;
  const __bf16* hp = h + (size_t)m * Cc;
  const float* ap = lA + (size_t)rr * Cc;
  float s = 0.f;
  for (int k = 0; k < Cc; ++k) s += (float)hp[k] * ap[k];
  t[i] = s;
}

// ---------------- qkv finalize: add LoRA, split, reshape, convert to bf16 -------
// q,k: [B,H,T,64] bf16 (q pre-scaled by 1/sqrt(d)); vT: [B,H,64,T] bf16
__global__ void qkv_split_kernel(__bf16* __restrict__ qo, __bf16* __restrict__ ko,
                                 __bf16* __restrict__ vto,
                                 const float* __restrict__ qkv,
                                 const float* __restrict__ t,
                                 const float* __restrict__ lB) {
  int i = blockIdx.x * blockDim.x + threadIdx.x;
  if (i >= Mc * 3 * Cc) return;
  int m = i / (3 * Cc), n = i % (3 * Cc);
  float acc = qkv[i];
  const float* tb = t + (size_t)m * Rc;
  const float* lb = lB + (size_t)n * Rc;
#pragma unroll
  for (int rr = 0; rr < Rc; ++rr) acc += LORA_SCALE * tb[rr] * lb[rr];
  int part = n / Cc, c = n % Cc, h = c / Dh, d = c % Dh;
  int b = m / Tc, tq = m % Tc;
  size_t base = (((size_t)b * Hc + h) * Tc + tq) * Dh + d;
  if (part == 0)      qo[base] = (__bf16)(acc * QSCALE);
  else if (part == 1) ko[base] = (__bf16)acc;
  else vto[(((size_t)b * Hc + h) * Dh + d) * Tc + tq] = (__bf16)acc;
}

// ---------------- flash attention: one wave per (b, h, 16-row q tile) -----------
__global__ __launch_bounds__(32) void attn_kernel(__bf16* __restrict__ y,
                                                  const __bf16* __restrict__ q,
                                                  const __bf16* __restrict__ k,
                                                  const __bf16* __restrict__ vt) {
  __shared__ __attribute__((aligned(16))) __bf16 plds[16 * 32];
  const int lane = threadIdx.x;
  const int r = lane & 15, sel = lane >> 4;
  const int qt = blockIdx.x % (Tc / 16);
  const int h = (blockIdx.x / (Tc / 16)) % Hc;
  const int b = blockIdx.x / ((Tc / 16) * Hc);
  const int q0 = qt * 16;

  // load Q as two A-fragments covering d = 0..31 and 32..63
  const __bf16* qr = q + (((size_t)b * Hc + h) * Tc + q0 + r) * Dh;
  BFrag qf[2];
#pragma unroll
  for (int s = 0; s < 2; ++s) {
    int d0 = s * 32;
    qf[s].q[0] = *(const uint4*)(qr + d0 + sel * 8);
    qf[s].q[1] = *(const uint4*)(qr + d0 + 16 + sel * 8);
  }

  float mrow[8], lrow[8];
#pragma unroll
  for (int v = 0; v < 8; ++v) { mrow[v] = -3.0e38f; lrow[v] = 0.f; }
  v8f o[4] = {zero8(), zero8(), zero8(), zero8()};

  const __bf16* kbase = k + ((size_t)b * Hc + h) * Tc * Dh;
  const __bf16* vbase = vt + ((size_t)b * Hc + h) * Dh * Tc;

  for (int kt0 = 0; kt0 < q0 + 16; kt0 += 32) {
    // scores for two 16-key sub-tiles
    v8f s[2] = {zero8(), zero8()};
#pragma unroll
    for (int sub = 0; sub < 2; ++sub) {
      const __bf16* kr = kbase + (size_t)(kt0 + sub * 16 + r) * Dh;
#pragma unroll
      for (int ks = 0; ks < 2; ++ks) {
        int d0 = ks * 32;
        BFrag kf;
        kf.q[0] = *(const uint4*)(kr + d0 + sel * 8);
        kf.q[1] = *(const uint4*)(kr + d0 + 16 + sel * 8);
        s[sub] = wmma_bf16(qf[ks], kf, s[sub]);
      }
    }
    // causal mask + online softmax
    float p0[8], p1[8], bm[8], fac[8], rs[8];
#pragma unroll
    for (int v = 0; v < 8; ++v) {
      int qidx = q0 + sel * 8 + v;
      float s0 = (kt0 + r > qidx) ? -1.0e30f : s[0][v];
      float s1 = (kt0 + 16 + r > qidx) ? -1.0e30f : s[1][v];
      s[0][v] = s0; s[1][v] = s1;
      bm[v] = fmaxf(s0, s1);
    }
#pragma unroll
    for (int v = 0; v < 8; ++v)
      for (int off = 1; off <= 8; off <<= 1)
        bm[v] = fmaxf(bm[v], __shfl_xor(bm[v], off, 32));
#pragma unroll
    for (int v = 0; v < 8; ++v) {
      float mn = fmaxf(mrow[v], bm[v]);
      fac[v] = __expf(mrow[v] - mn);
      mrow[v] = mn;
      p0[v] = __expf(s[0][v] - mn);
      p1[v] = __expf(s[1][v] - mn);
      rs[v] = p0[v] + p1[v];
    }
#pragma unroll
    for (int v = 0; v < 8; ++v)
      for (int off = 1; off <= 8; off <<= 1)
        rs[v] += __shfl_xor(rs[v], off, 32);
#pragma unroll
    for (int v = 0; v < 8; ++v) lrow[v] = lrow[v] * fac[v] + rs[v];
#pragma unroll
    for (int dn = 0; dn < 4; ++dn)
#pragma unroll
      for (int v = 0; v < 8; ++v) o[dn][v] *= fac[v];

    // restage P (C layout -> A layout) through LDS as bf16
    __syncthreads();
#pragma unroll
    for (int v = 0; v < 8; ++v) {
      int m = sel * 8 + v;
      plds[m * 32 + r]      = (__bf16)p0[v];
      plds[m * 32 + 16 + r] = (__bf16)p1[v];
    }
    __syncthreads();
    BFrag pf;
    pf.q[0] = *(const uint4*)&plds[r * 32 + sel * 8];
    pf.q[1] = *(const uint4*)&plds[r * 32 + 16 + sel * 8];

    // O += P(16x32) . V(32x64), V^T rows are contiguous in keys
#pragma unroll
    for (int dn = 0; dn < 4; ++dn) {
      const __bf16* vr = vbase + (size_t)(dn * 16 + r) * Tc + kt0;
      BFrag vf;
      vf.q[0] = *(const uint4*)(vr + sel * 8);
      vf.q[1] = *(const uint4*)(vr + 16 + sel * 8);
      o[dn] = wmma_bf16(pf, vf, o[dn]);
    }
  }

  // epilogue: y[b*T+q, h*64+d] = O / l  (bf16)
  float inv[8];
#pragma unroll
  for (int v = 0; v < 8; ++v) inv[v] = 1.0f / lrow[v];
#pragma unroll
  for (int dn = 0; dn < 4; ++dn) {
#pragma unroll
    for (int v = 0; v < 8; ++v) {
      int m = sel * 8 + v;
      y[((size_t)b * Tc + q0 + m) * Cc + h * Dh + dn * 16 + r] =
          (__bf16)(o[dn][v] * inv[v]);
    }
  }
}

// ---------------- residual add (+ optional LoRA term): x += y + s*t.lB^T --------
__global__ void add_res_kernel(float* __restrict__ x, const float* __restrict__ yv,
                               const float* __restrict__ t,
                               const float* __restrict__ lB, int N) {
  int i = blockIdx.x * blockDim.x + threadIdx.x;
  if (i >= Mc * N) return;
  int m = i / N, n = i % N;
  float acc = x[i] + yv[i];
  if (t) {
    const float* tb = t + (size_t)m * Rc;
    const float* lb = lB + (size_t)n * Rc;
#pragma unroll
    for (int rr = 0; rr < Rc; ++rr) acc += LORA_SCALE * tb[rr] * lb[rr];
  }
  x[i] = acc;
}

// ---------------- GELU (tanh approx), f32 in -> bf16 out ----------------
__global__ void gelu_kernel(__bf16* __restrict__ out, const float* __restrict__ in,
                            int n) {
  int i = blockIdx.x * blockDim.x + threadIdx.x;
  if (i >= n) return;
  float u = in[i];
  float tg = tanhf(0.7978845608028654f * (u + 0.044715f * u * u * u));
  out[i] = (__bf16)(0.5f * u * (1.0f + tg));
}

// =====================================================================
extern "C" void kernel_launch(void* const* d_in, const int* in_sizes, int n_in,
                              void* d_out, int out_size, void* d_ws, size_t ws_size,
                              hipStream_t stream) {
  (void)in_sizes; (void)n_in; (void)out_size; (void)ws_size;
  const int* idx       = (const int*)d_in[0];
  const float* wte     = (const float*)d_in[1];
  const float* wpe     = (const float*)d_in[2];
  const float* ln1_g   = (const float*)d_in[3];
  const float* ln1_b   = (const float*)d_in[4];
  const float* attn_w  = (const float*)d_in[5];
  const float* attn_lA = (const float*)d_in[6];
  const float* attn_lB = (const float*)d_in[7];
  const float* proj_w  = (const float*)d_in[8];
  const float* proj_lA = (const float*)d_in[9];
  const float* proj_lB = (const float*)d_in[10];
  const float* ln2_g   = (const float*)d_in[11];
  const float* ln2_b   = (const float*)d_in[12];
  const float* fc_w    = (const float*)d_in[13];
  const float* fc_b    = (const float*)d_in[14];
  const float* mproj_w = (const float*)d_in[15];
  const float* mproj_b = (const float*)d_in[16];
  const float* lnf_g   = (const float*)d_in[17];
  const float* lnf_b   = (const float*)d_in[18];
  const float* head_w  = (const float*)d_in[19];
  float* logits = (float*)d_out;

  // ---- bump allocator over workspace ----
  size_t off = 0;
  auto alloc = [&](size_t bytes) -> char* {
    char* p = (char*)d_ws + off;
    off += (bytes + 255) & ~(size_t)255;
    return p;
  };
  __bf16* attn_w_bf  = (__bf16*)alloc((size_t)Lc * 3 * Cc * Cc * 2);
  __bf16* proj_w_bf  = (__bf16*)alloc((size_t)Lc * Cc * Cc * 2);
  __bf16* fc_w_bf    = (__bf16*)alloc((size_t)Lc * 4 * Cc * Cc * 2);
  __bf16* mproj_w_bf = (__bf16*)alloc((size_t)Lc * Cc * 4 * Cc * 2);
  __bf16* head_w_bf  = (__bf16*)alloc((size_t)Vpad * Cc * 2);  // zero-padded rows
  float* x    = (float*)alloc((size_t)Mc * Cc * 4);
  float* qkv  = (float*)alloc((size_t)Mc * 3 * Cc * 4);
  float* fco  = (float*)alloc((size_t)Mc * 4 * Cc * 4);
  float* yg   = (float*)alloc((size_t)Mc * Cc * 4);
  float* t1   = (float*)alloc((size_t)Mc * Rc * 4);
  float* t2   = (float*)alloc((size_t)Mc * Rc * 4);
  __bf16* hbf = (__bf16*)alloc((size_t)Mc * Cc * 2);
  __bf16* qb  = (__bf16*)alloc((size_t)Bc * Hc * Tc * Dh * 2);
  __bf16* kb  = (__bf16*)alloc((size_t)Bc * Hc * Tc * Dh * 2);
  __bf16* vtb = (__bf16*)alloc((size_t)Bc * Hc * Dh * Tc * 2);
  __bf16* yab = (__bf16*)alloc((size_t)Mc * Cc * 2);
  __bf16* glb = (__bf16*)alloc((size_t)Mc * 4 * Cc * 2);

  auto cvt = [&](__bf16* dst, const float* src, size_t n, size_t npad) {
    int nn = (int)n, np = (int)npad;
    cvt_bf16_kernel<<<(np + 255) / 256, 256, 0, stream>>>(dst, src, nn, np);
  };
  auto gemm = [&](float* out, const __bf16* A, const __bf16* W,
                  const float* bias, int M, int N, int K, int nt) {
    dim3 g(((N + 63) / 64), M / 128);
    wmma_gemm_kernel<<<g, 128, 0, stream>>>(out, A, W, bias, M, N, K, nt);
  };

  // weights -> bf16 (once per call); head padded with zero rows to Vpad
  cvt(attn_w_bf, attn_w, (size_t)Lc * 3 * Cc * Cc, (size_t)Lc * 3 * Cc * Cc);
  cvt(proj_w_bf, proj_w, (size_t)Lc * Cc * Cc, (size_t)Lc * Cc * Cc);
  cvt(fc_w_bf, fc_w, (size_t)Lc * 4 * Cc * Cc, (size_t)Lc * 4 * Cc * Cc);
  cvt(mproj_w_bf, mproj_w, (size_t)Lc * Cc * 4 * Cc, (size_t)Lc * Cc * 4 * Cc);
  cvt(head_w_bf, head_w, (size_t)Vc * Cc, (size_t)Vpad * Cc);

  // embedding
  embed_kernel<<<(Mc * Cc + 255) / 256, 256, 0, stream>>>(x, idx, wte, wpe);

  for (int i = 0; i < Lc; ++i) {
    // ---- attention ----
    ln_kernel<<<Mc, 256, 0, stream>>>(hbf, x, ln1_g + i * Cc, ln1_b + i * Cc);
    gemm(qkv, hbf, attn_w_bf + (size_t)i * 3 * Cc * Cc, nullptr, Mc, 3 * Cc, Cc, 0);
    lora_t_kernel<<<(Mc * Rc + 255) / 256, 256, 0, stream>>>(
        t1, hbf, attn_lA + (size_t)i * Rc * Cc);
    qkv_split_kernel<<<(Mc * 3 * Cc + 255) / 256, 256, 0, stream>>>(
        qb, kb, vtb, qkv, t1, attn_lB + (size_t)i * 3 * Cc * Rc);
    attn_kernel<<<Bc * Hc * (Tc / 16), 32, 0, stream>>>(yab, qb, kb, vtb);
    gemm(yg, yab, proj_w_bf + (size_t)i * Cc * Cc, nullptr, Mc, Cc, Cc, 0);
    lora_t_kernel<<<(Mc * Rc + 255) / 256, 256, 0, stream>>>(
        t2, yab, proj_lA + (size_t)i * Rc * Cc);
    add_res_kernel<<<(Mc * Cc + 255) / 256, 256, 0, stream>>>(
        x, yg, t2, proj_lB + (size_t)i * Cc * Rc, Cc);
    // ---- MLP ----
    ln_kernel<<<Mc, 256, 0, stream>>>(hbf, x, ln2_g + i * Cc, ln2_b + i * Cc);
    gemm(fco, hbf, fc_w_bf + (size_t)i * 4 * Cc * Cc, fc_b + (size_t)i * 4 * Cc,
         Mc, 4 * Cc, Cc, 0);
    gelu_kernel<<<(Mc * 4 * Cc + 255) / 256, 256, 0, stream>>>(glb, fco,
                                                               Mc * 4 * Cc);
    gemm(yg, glb, mproj_w_bf + (size_t)i * Cc * 4 * Cc,
         mproj_b + (size_t)i * Cc, Mc, Cc, 4 * Cc, 0);
    add_res_kernel<<<(Mc * Cc + 255) / 256, 256, 0, stream>>>(
        x, yg, nullptr, nullptr, Cc);
  }

  // final LN + LM head (2048 x 50257 x 512), streaming NT stores for logits
  ln_kernel<<<Mc, 256, 0, stream>>>(hbf, x, lnf_g, lnf_b);
  gemm(logits, hbf, head_w_bf, nullptr, Mc, Vc, Cc, 1);
}
